// GIKTModel_12421045420330
// MI455X (gfx1250) — matrix operand order repaired and checked by reading, not compiled, not run on hardware
//
#include <hip/hip_runtime.h>
#include <hip/hip_bf16.h>
#include <math.h>

typedef __attribute__((ext_vector_type(16))) __bf16 v16bf;
typedef __attribute__((ext_vector_type(8)))  __bf16 v8bf;
typedef __attribute__((ext_vector_type(8)))  float  v8f;
typedef __attribute__((ext_vector_type(4)))  float  f32x4;

static __device__ __forceinline__ float sigf(float x) {
    return 1.0f / (1.0f + __expf(-x));
}

#define WMMA_BF16(A_, B_, C_) \
    __builtin_amdgcn_wmma_f32_16x16x32_bf16(false, (A_), false, (B_), (short)0, (C_), false, false)

// ---------------------------------------------------------------------------
// B-fragment packing (32 contiguous bytes per lane per fragment):
// fragment f = nt*(K/32)+kt; lane l holds n = nt*16+(l&15),
// k = kt*32 + (l>>4)*8 + i + (i&8), i=0..15.
// ---------------------------------------------------------------------------
__global__ void pack_KN(const float* __restrict__ B, __bf16* __restrict__ out,
                        int K, int N) {
    int t = blockIdx.x * blockDim.x + threadIdx.x;
    int total = (N / 16) * (K / 32) * 32;
    if (t >= total) return;
    int lane = t & 31;
    int frag = t >> 5;
    int KT = K / 32;
    int kt = frag % KT;
    int nt = frag / KT;
    int n = nt * 16 + (lane & 15);
    int kbase = kt * 32 + (lane >> 4) * 8;
    __bf16* dst = out + (size_t)t * 16;
#pragma unroll
    for (int i = 0; i < 16; ++i)
        dst[i] = (__bf16)B[(size_t)(kbase + i + (i & 8)) * N + n];
}

// LSTM weights: Wih/Whh [1024,256] row-major ([N,Kpart]).
// Logical B[k,n] (K=512,N=1024): k<256 -> Wih[n,k], else Whh[n,k-256].
__global__ void pack_lstm(const float* __restrict__ Wih,
                          const float* __restrict__ Whh,
                          __bf16* __restrict__ out) {
    const int KT = 16;
    int t = blockIdx.x * blockDim.x + threadIdx.x;
    int total = (1024 / 16) * KT * 32;
    if (t >= total) return;
    int lane = t & 31;
    int frag = t >> 5;
    int kt = frag % KT;
    int nt = frag / KT;
    int n = nt * 16 + (lane & 15);
    int kbase = kt * 32 + (lane >> 4) * 8;
    __bf16* dst = out + (size_t)t * 16;
#pragma unroll
    for (int i = 0; i < 16; ++i) {
        int k = kbase + i + (i & 8);
        float v = (k < 256) ? Wih[(size_t)n * 256 + k]
                            : Whh[(size_t)n * 256 + (k - 256)];
        dst[i] = (__bf16)v;
    }
}

// ---------------------------------------------------------------------------
// Barrier-free WMMA GEMM, bf16 A (natural row-major), packed B.
// Output: f32 (Cf) or bf16 (Cb) — exactly one non-null.
// block = 128 (4 waves); wave = 2 row-fragments x 16 cols. grid = (N/64, M/32).
// ---------------------------------------------------------------------------
__global__ void gemm_abf(const __bf16* __restrict__ A, int lda,
                         const __bf16* __restrict__ Bpack,
                         const float* __restrict__ bias,
                         float* __restrict__ Cf, __bf16* __restrict__ Cb,
                         int N, int K, int act) {
    int lane = threadIdx.x & 31;
    int w = threadIdx.x >> 5;
    int r0 = blockIdx.y * 32;
    int c0 = blockIdx.x * 64 + w * 16;
    int nt = c0 >> 4;
    int KT = K >> 5;
    int ml = lane & 15;
    int ka = (lane >> 4) * 8;

    const __bf16* a0p = A + (size_t)(r0 + ml) * lda + ka;
    const __bf16* a1p = A + (size_t)(r0 + 16 + ml) * lda + ka;
    const __bf16* bp  = Bpack + ((size_t)nt * KT * 32 + lane) * 16;

    v8f acc0 = {}, acc1 = {};
#pragma unroll 4
    for (int kt = 0; kt < KT; ++kt) {
        v8bf l0 = *(const v8bf*)(a0p + kt * 32);
        v8bf m0 = *(const v8bf*)(a0p + kt * 32 + 16);
        v8bf l1 = *(const v8bf*)(a1p + kt * 32);
        v8bf m1 = *(const v8bf*)(a1p + kt * 32 + 16);
        v16bf a0, a1;
#pragma unroll
        for (int i = 0; i < 8; ++i) {
            a0[i] = l0[i]; a0[8 + i] = m0[i];
            a1[i] = l1[i]; a1[8 + i] = m1[i];
        }
        v16bf b = *(const v16bf*)(bp + kt * 512);
        acc0 = WMMA_BF16(a0, b, acc0);
        acc1 = WMMA_BF16(a1, b, acc1);
    }

    int col = c0 + ml;
    float bv = bias ? bias[col] : 0.0f;
    int rb = r0 + ((lane >> 4) * 8);
#pragma unroll
    for (int p = 0; p < 8; ++p) {
        float v0 = acc0[p] + bv;
        float v1 = acc1[p] + bv;
        if (act == 1) { v0 = fmaxf(v0, 0.0f); v1 = fmaxf(v1, 0.0f); }
        if (Cf) {
            Cf[(size_t)(rb + p) * N + col]      = v0;
            Cf[(size_t)(rb + 16 + p) * N + col] = v1;
        } else {
            Cb[(size_t)(rb + p) * N + col]      = (__bf16)v0;
            Cb[(size_t)(rb + 16 + p) * N + col] = (__bf16)v1;
        }
    }
}

// ---------------------------------------------------------------------------
// Persistent fused LSTM: all 200 steps in one launch.
//  - cell state in VGPRs for the whole sequence
//  - x_{t+1} cooperatively converted into A-fragment-packed bf16 during step t
//    (doubles as prefetch); h stored fragment-packed for 32B-contiguous reads
//  - inner loop: per K-chunk, 6 contiguous 32B loads + 8 WMMAs; zero cvt
//  - cross-WGP sync: device-scope atomic release-arrive / acquire-spin
// block = 128 (4 waves), grid = (4, 32) -> 128 WGs = 16384 threads.
// ---------------------------------------------------------------------------
#define LSTM_WGS 128u

// Convert one lane-fragment of x at timestep base Xt into packed layout.
// tid in [0, 16384): frag = tid>>5 covers (rt = tid>>8, kt = (tid>>5)&7).
// A-frag layout: lane l holds rows rt*16+(l&15), k = kt*32 + (l>>4)*8 + {0..7}
// then +16.
static __device__ __forceinline__ void xconvert(const float* __restrict__ Xt,
                                                __bf16* __restrict__ dst,
                                                int tid, int SH) {
    int rt  = tid >> 8;
    int rem = tid & 255;
    int kt  = rem >> 5;
    int l   = rem & 31;
    int row = rt * 16 + (l & 15);
    int k0  = kt * 32 + ((l >> 4) * 8);
    const float* s = Xt + (size_t)row * SH + k0;
    f32x4 u0 = *(const f32x4*)(s);
    f32x4 u1 = *(const f32x4*)(s + 4);
    f32x4 u2 = *(const f32x4*)(s + 16);
    f32x4 u3 = *(const f32x4*)(s + 20);
    v16bf o;
#pragma unroll
    for (int i = 0; i < 4; ++i) {
        o[i]      = (__bf16)u0[i]; o[4 + i]  = (__bf16)u1[i];
        o[8 + i]  = (__bf16)u2[i]; o[12 + i] = (__bf16)u3[i];
    }
    *(v16bf*)(dst + (size_t)tid * 16) = o;
}

__global__ void lstm_persistent(const float* __restrict__ X, int S,
                                __bf16* __restrict__ xp0,
                                __bf16* __restrict__ xp1,
                                __bf16* __restrict__ hp0,
                                __bf16* __restrict__ hp1,
                                const __bf16* __restrict__ Bpack,
                                const float* __restrict__ bih,
                                const float* __restrict__ bhh,
                                __bf16* __restrict__ hnat,
                                unsigned* __restrict__ bar) {
    const int H = 256, KT = 16;  // K = 512 (256 x-part + 256 h-part)
    const int SH = S * H;
    int lane = threadIdx.x & 31;
    int w = threadIdx.x >> 5;
    int r0 = blockIdx.y * 32;
    int rt0 = blockIdx.y * 2;
    int j0 = blockIdx.x * 64 + w * 16;
    int nti = j0 >> 4;
    int ml = lane & 15;
    int tid = (blockIdx.y * 4 + blockIdx.x) * 128 + threadIdx.x;  // 0..16383

    const __bf16* bp0 = Bpack + ((size_t)(nti     ) * KT * 32 + lane) * 16;
    const __bf16* bp1 = Bpack + ((size_t)(nti + 16) * KT * 32 + lane) * 16;
    const __bf16* bp2 = Bpack + ((size_t)(nti + 32) * KT * 32 + lane) * 16;
    const __bf16* bp3 = Bpack + ((size_t)(nti + 48) * KT * 32 + lane) * 16;

    int col = j0 + ml;
    float bi = bih[col]       + bhh[col];
    float bf = bih[col + 256] + bhh[col + 256];
    float bg = bih[col + 512] + bhh[col + 512];
    float bo = bih[col + 768] + bhh[col + 768];
    int rb = r0 + ((lane >> 4) * 8);

    // fragment-packed store coordinates for h (col is the k-dim of next step)
    int kk  = col & 31;
    int ct  = col >> 5;
    int khi = (kk >> 3) & 1;                       // lane' bit 4
    int ei  = (kk & 7) | ((kk & 16) >> 1);         // element index in fragment
    int lpb = (lane >> 4) * 8 + (khi << 4);        // + p -> lane' in fragment
    size_t f0 = ((size_t)(rt0 * 8 + ct) * 32) * 16 + ei;
    size_t f1 = ((size_t)((rt0 + 1) * 8 + ct) * 32) * 16 + ei;

    // cell state lives in registers for the entire sequence
    float c0r[8], c1r[8];
#pragma unroll
    for (int p = 0; p < 8; ++p) { c0r[p] = 0.0f; c1r[p] = 0.0f; }

    unsigned phase = 0;

    // prologue: convert x_0 into xp0, then grid-sync
    xconvert(X, xp0, tid, SH);
    {
        __threadfence();
        __syncthreads();
        ++phase;
        if (threadIdx.x == 0) {
            __hip_atomic_fetch_add(bar, 1u, __ATOMIC_RELEASE,
                                   __HIP_MEMORY_SCOPE_AGENT);
            while (__hip_atomic_load(bar, __ATOMIC_ACQUIRE,
                                     __HIP_MEMORY_SCOPE_AGENT) < LSTM_WGS * phase)
                __builtin_amdgcn_s_sleep(2);
        }
        __syncthreads();
    }

    for (int t = 0; t < S; ++t) {
        const __bf16* xc  = (t & 1) ? xp1 : xp0;       // x_t (packed)
        const __bf16* hc  = (t & 1) ? hp1 : hp0;       // h_t (packed, t>0)
        __bf16*       hnx = (t & 1) ? hp0 : hp1;       // h_{t+1} (packed)

        const __bf16* xa0 = xc + ((size_t)(rt0    ) * 256 + lane) * 16;
        const __bf16* xa1 = xc + ((size_t)(rt0 + 1) * 256 + lane) * 16;
        const __bf16* ha0 = hc + ((size_t)(rt0    ) * 256 + lane) * 16;
        const __bf16* ha1 = hc + ((size_t)(rt0 + 1) * 256 + lane) * 16;

        v8f a0i = {}, a0f = {}, a0g = {}, a0o = {};
        v8f a1i = {}, a1f = {}, a1g = {}, a1o = {};

        // ---- x-part of K: 2 A-frag + 4 B-frag 32B loads, 8 WMMAs per chunk
#pragma unroll
        for (int kt = 0; kt < 8; ++kt) {
            v16bf a0 = *(const v16bf*)(xa0 + kt * 512);
            v16bf a1 = *(const v16bf*)(xa1 + kt * 512);
            v16bf b0 = *(const v16bf*)(bp0 + kt * 512);
            v16bf b1 = *(const v16bf*)(bp1 + kt * 512);
            v16bf b2 = *(const v16bf*)(bp2 + kt * 512);
            v16bf b3 = *(const v16bf*)(bp3 + kt * 512);
            a0i = WMMA_BF16(a0, b0, a0i);  a0f = WMMA_BF16(a0, b1, a0f);
            a0g = WMMA_BF16(a0, b2, a0g);  a0o = WMMA_BF16(a0, b3, a0o);
            a1i = WMMA_BF16(a1, b0, a1i);  a1f = WMMA_BF16(a1, b1, a1f);
            a1g = WMMA_BF16(a1, b2, a1g);  a1o = WMMA_BF16(a1, b3, a1o);
        }

        // ---- h-part of K (h == 0 at t == 0, skip) ----
        if (t > 0) {
#pragma unroll
            for (int kt = 0; kt < 8; ++kt) {
                v16bf a0 = *(const v16bf*)(ha0 + kt * 512);
                v16bf a1 = *(const v16bf*)(ha1 + kt * 512);
                v16bf b0 = *(const v16bf*)(bp0 + (kt + 8) * 512);
                v16bf b1 = *(const v16bf*)(bp1 + (kt + 8) * 512);
                v16bf b2 = *(const v16bf*)(bp2 + (kt + 8) * 512);
                v16bf b3 = *(const v16bf*)(bp3 + (kt + 8) * 512);
                a0i = WMMA_BF16(a0, b0, a0i);  a0f = WMMA_BF16(a0, b1, a0f);
                a0g = WMMA_BF16(a0, b2, a0g);  a0o = WMMA_BF16(a0, b3, a0o);
                a1i = WMMA_BF16(a1, b0, a1i);  a1f = WMMA_BF16(a1, b1, a1f);
                a1g = WMMA_BF16(a1, b2, a1g);  a1o = WMMA_BF16(a1, b3, a1o);
            }
        }

        // ---- convert x_{t+1} into the other ping-pong buffer (prefetch) ----
        if (t + 1 < S)
            xconvert(X + (size_t)(t + 1) * H, (t & 1) ? xp0 : xp1, tid, SH);

        // ---- fused gates + register-resident cell update ----
#pragma unroll
        for (int p = 0; p < 8; ++p) {
            {
                float gi = sigf(a0i[p] + bi);
                float gf = sigf(a0f[p] + bf);
                float gg = tanhf(a0g[p] + bg);
                float go = sigf(a0o[p] + bo);
                float cn = gf * c0r[p] + gi * gg;
                c0r[p] = cn;
                float hv = go * tanhf(cn);
                hnx[f0 + (size_t)(lpb + p) * 16] = (__bf16)hv;
                if (t == S - 1) hnat[(size_t)(rb + p) * H + col] = (__bf16)hv;
            }
            {
                float gi = sigf(a1i[p] + bi);
                float gf = sigf(a1f[p] + bf);
                float gg = tanhf(a1g[p] + bg);
                float go = sigf(a1o[p] + bo);
                float cn = gf * c1r[p] + gi * gg;
                c1r[p] = cn;
                float hv = go * tanhf(cn);
                hnx[f1 + (size_t)(lpb + p) * 16] = (__bf16)hv;
                if (t == S - 1) hnat[(size_t)(rb + 16 + p) * H + col] = (__bf16)hv;
            }
        }

        // ---- device-wide split barrier ----
        if (t + 1 < S) {
            __threadfence();
            __syncthreads();
            ++phase;
            if (threadIdx.x == 0) {
                __hip_atomic_fetch_add(bar, 1u, __ATOMIC_RELEASE,
                                       __HIP_MEMORY_SCOPE_AGENT);
                while (__hip_atomic_load(bar, __ATOMIC_ACQUIRE,
                                         __HIP_MEMORY_SCOPE_AGENT) < LSTM_WGS * phase)
                    __builtin_amdgcn_s_sleep(2);
            }
            __syncthreads();
        }
    }
}

// ---------------------------------------------------------------------------
// Small helpers
// ---------------------------------------------------------------------------
__global__ void gather_concat(const int* __restrict__ sid,
                              const int* __restrict__ qid,
                              const int* __restrict__ kid,
                              const float* __restrict__ semb,
                              const float* __restrict__ qemb,
                              const float* __restrict__ kemb,
                              __bf16* __restrict__ out) {
    int t = blockIdx.x * blockDim.x + threadIdx.x;  // 1024*384
    if (t >= 1024 * 384) return;
    int b = t / 384, f = t % 384;
    float v;
    if (f < 128)       v = semb[(size_t)sid[b] * 128 + f];
    else if (f < 256)  v = qemb[(size_t)qid[b] * 128 + (f - 128)];
    else               v = kemb[(size_t)kid[b] * 128 + (f - 256)];
    out[t] = (__bf16)v;
}

__global__ void concat2(const __bf16* __restrict__ a,
                        const __bf16* __restrict__ b,
                        __bf16* __restrict__ out) {
    int t = blockIdx.x * blockDim.x + threadIdx.x;  // 1024*512
    if (t >= 1024 * 512) return;
    int r = t >> 9, f = t & 511;
    out[t] = (f < 256) ? a[r * 256 + f] : b[r * 256 + (f - 256)];
}

__global__ void zero_u32(unsigned* p, int n) {
    int t = blockIdx.x * blockDim.x + threadIdx.x;
    if (t < n) p[t] = 0u;
}

// out[b] = sigmoid(dot(h2[b,:], W[:,0]) + b2). One wave per row.
__global__ void pred2_kernel(const float* __restrict__ h2,
                             const float* __restrict__ W,
                             const float* __restrict__ b2,
                             float* __restrict__ out) {
    int lane = threadIdx.x & 31;
    int row = blockIdx.x * 8 + (threadIdx.x >> 5);
    float s = 0.0f;
    for (int k = lane; k < 256; k += 32)
        s += h2[(size_t)row * 256 + k] * W[k];
#pragma unroll
    for (int off = 16; off > 0; off >>= 1)
        s += __shfl_xor(s, off, 32);
    if (lane == 0) out[row] = sigf(s + b2[0]);
}

// ---------------------------------------------------------------------------
extern "C" void kernel_launch(void* const* d_in, const int* in_sizes, int n_in,
                              void* d_out, int out_size, void* d_ws, size_t ws_size,
                              hipStream_t stream) {
    (void)in_sizes; (void)n_in; (void)out_size; (void)ws_size;
    const int*   sid  = (const int*)d_in[0];
    const int*   qid  = (const int*)d_in[1];
    const int*   kid  = (const int*)d_in[2];
    // d_in[3] graph_x, d_in[4] edge_index: GCN output is dead code -> skipped
    const float* X    = (const float*)d_in[5];   // [1024, 200, 256]
    const float* semb = (const float*)d_in[6];
    const float* qemb = (const float*)d_in[7];
    const float* kemb = (const float*)d_in[8];
    const float* iW1  = (const float*)d_in[13];  // [384, 256]
    const float* ib1  = (const float*)d_in[14];
    const float* iW2  = (const float*)d_in[15];  // [256, 256]
    const float* ib2  = (const float*)d_in[16];
    const float* pW1  = (const float*)d_in[17];  // [512, 256]
    const float* pb1  = (const float*)d_in[18];
    const float* pW2  = (const float*)d_in[19];  // [256, 1]
    const float* pb2  = (const float*)d_in[20];
    const float* Wih  = (const float*)d_in[21];  // [1024, 256]
    const float* Whh  = (const float*)d_in[22];  // [1024, 256]
    const float* bih  = (const float*)d_in[23];
    const float* bhh  = (const float*)d_in[24];

    // workspace carving (256B aligned)
    char* ws = (char*)d_ws;
    size_t o = 0;
    auto carve = [&](size_t bytes) -> char* {
        char* p = ws + o;
        o += (bytes + 255) & ~(size_t)255;
        return p;
    };
    __bf16*   bpL   = (__bf16*)carve(512 * 1024 * 2);  // LSTM packed weights
    __bf16*   bp1   = (__bf16*)carve(384 * 256 * 2);
    __bf16*   bp2   = (__bf16*)carve(256 * 256 * 2);
    __bf16*   bpP   = (__bf16*)carve(512 * 256 * 2);
    __bf16*   xpk0  = (__bf16*)carve(1024 * 256 * 2);  // packed x ping-pong
    __bf16*   xpk1  = (__bf16*)carve(1024 * 256 * 2);
    __bf16*   hpk0  = (__bf16*)carve(1024 * 256 * 2);  // packed h ping-pong
    __bf16*   hpk1  = (__bf16*)carve(1024 * 256 * 2);
    __bf16*   hnat  = (__bf16*)carve(1024 * 256 * 2);  // h_last, natural layout
    __bf16*   ainb  = (__bf16*)carve(1024 * 384 * 2);
    __bf16*   int1b = (__bf16*)carve(1024 * 256 * 2);
    __bf16*   int2b = (__bf16*)carve(1024 * 256 * 2);
    __bf16*   combb = (__bf16*)carve(1024 * 512 * 2);
    float*    h2    = (float*)carve(1024 * 256 * 4);
    unsigned* bar   = (unsigned*)carve(256);

    // pack weights into WMMA fragment layout (L2-resident thereafter)
    pack_lstm<<<(64 * 16 * 32 + 255) / 256, 256, 0, stream>>>(Wih, Whh, bpL);
    pack_KN<<<(16 * 12 * 32 + 255) / 256, 256, 0, stream>>>(iW1, bp1, 384, 256);
    pack_KN<<<(16 * 8  * 32 + 255) / 256, 256, 0, stream>>>(iW2, bp2, 256, 256);
    pack_KN<<<(16 * 16 * 32 + 255) / 256, 256, 0, stream>>>(pW1, bpP, 512, 256);

    // reset the grid-barrier counter (ws is not re-initialized by the harness)
    zero_u32<<<1, 64, 0, stream>>>(bar, 64);

    // interaction MLP (bf16 activations end to end)
    gather_concat<<<(1024 * 384 + 255) / 256, 256, 0, stream>>>(
        sid, qid, kid, semb, qemb, kemb, ainb);
    gemm_abf<<<dim3(4, 32), 128, 0, stream>>>(ainb, 384, bp1, ib1,
                                              nullptr, int1b, 256, 384, 1);
    gemm_abf<<<dim3(4, 32), 128, 0, stream>>>(int1b, 256, bp2, ib2,
                                              nullptr, int2b, 256, 256, 1);

    // LSTM: one persistent launch, all 200 steps, grid-sync between steps
    lstm_persistent<<<dim3(4, 32), 128, 0, stream>>>(
        X, 200, xpk0, xpk1, hpk0, hpk1, bpL, bih, bhh, hnat, bar);

    // prediction head
    concat2<<<(1024 * 512 + 255) / 256, 256, 0, stream>>>(int2b, hnat, combb);
    gemm_abf<<<dim3(4, 32), 128, 0, stream>>>(combb, 512, bpP, pb1,
                                              h2, nullptr, 256, 512, 1);
    pred2_kernel<<<128, 256, 0, stream>>>(h2, pW2, pb2, (float*)d_out);
}
